// BlockchainAIEngine_2027224564539
// MI455X (gfx1250) — matrix operand order repaired and checked by reading, not compiled, not run on hardware
//
#include <hip/hip_runtime.h>
#include <hip/hip_bf16.h>

typedef __bf16 bf16_t;
typedef __attribute__((ext_vector_type(16))) __bf16 v16bf;
typedef __attribute__((ext_vector_type(8)))  __bf16 v8bf;
typedef __attribute__((ext_vector_type(4)))  __bf16 v4bf;
typedef __attribute__((ext_vector_type(8)))  float  v8f;
typedef __attribute__((ext_vector_type(4)))  int    v4i;

#define TPB 256

#if defined(__has_builtin)
#if __has_builtin(__builtin_amdgcn_global_load_async_to_lds_b128) && \
    __has_builtin(__builtin_amdgcn_s_wait_asynccnt)
#define USE_ASYNC_LDS 1
#endif
#endif
#ifndef USE_ASYNC_LDS
#define USE_ASYNC_LDS 0
#endif

#if USE_ASYNC_LDS
typedef __attribute__((address_space(1))) v4i global_v4i;   // HIP __device__ AS
typedef __attribute__((address_space(3))) v4i shared_v4i;   // HIP __shared__ AS
#endif

__device__ __forceinline__ void async_cp16(bf16_t* dst, const bf16_t* src) {
#if USE_ASYNC_LDS
    __builtin_amdgcn_global_load_async_to_lds_b128(
        (global_v4i*)src, (shared_v4i*)dst, /*offset=*/0, /*cpol=*/0);
#else
    *(v8bf*)dst = *(const v8bf*)src;
#endif
}

__device__ __forceinline__ void wait_async() {
#if USE_ASYNC_LDS
    __builtin_amdgcn_s_wait_asynccnt(0);
#endif
}

__device__ __forceinline__ v8f vzero8() {
    v8f z;
#pragma unroll
    for (int i = 0; i < 8; ++i) z[i] = 0.0f;
    return z;
}

__device__ __forceinline__ v8f wmma_bf16(v16bf a, v16bf b, v8f c) {
    return __builtin_amdgcn_wmma_f32_16x16x32_bf16(
        /*neg_a=*/false, a, /*neg_b=*/false, b,
        /*c_mod=*/(short)0, c, /*reuse_a=*/false, /*reuse_b=*/false);
}

// A fragment: 16x32 bf16, row-major LDS, row stride `stride` elems.
// lanes 0-15: row M=lane, K={0..7,16..23}; lanes 16-31: same rows, K={8..15,24..31}.
__device__ __forceinline__ v16bf load_a_frag(const bf16_t* __restrict__ base, int stride) {
    const int lane = threadIdx.x & 31;
    const int r    = lane & 15;
    const int kh   = (lane >> 4) << 3;
    const bf16_t* p = base + r * stride + kh;
    v8bf lo = *(const v8bf*)(p);
    v8bf hi = *(const v8bf*)(p + 16);
    v16bf a;
#pragma unroll
    for (int i = 0; i < 8; ++i) { a[i] = lo[i]; a[i + 8] = hi[i]; }
    return a;
}

// B fragment: 32x16 bf16 stored transposed [N][K] in LDS, K stride `stride`.
// lane n<16 -> col n, K={0..7,16..23}; lane 16+n -> col n, K={8..15,24..31}.
__device__ __forceinline__ v16bf load_b_frag(const bf16_t* __restrict__ baseT, int stride) {
    const int lane = threadIdx.x & 31;
    const int n    = lane & 15;
    const int kh   = (lane >> 4) << 3;
    const bf16_t* p = baseT + n * stride + kh;
    v8bf lo = *(const v8bf*)(p);
    v8bf hi = *(const v8bf*)(p + 16);
    v16bf b;
#pragma unroll
    for (int i = 0; i < 8; ++i) { b[i] = lo[i]; b[i + 8] = hi[i]; }
    return b;
}

// ---------------------------------------------------------------------------
// Pre-pass A: straight f32 -> bf16 convert (for x).
// ---------------------------------------------------------------------------
__global__ __launch_bounds__(TPB) void convert_kernel(
    const float* __restrict__ in, bf16_t* __restrict__ out, int n4) {
    int i = blockIdx.x * TPB + threadIdx.x;
    if (i < n4) {
        float4 v = ((const float4*)in)[i];
        v4bf o;
        o[0] = (bf16_t)v.x; o[1] = (bf16_t)v.y; o[2] = (bf16_t)v.z; o[3] = (bf16_t)v.w;
        ((v4bf*)out)[i] = o;
    }
}

// ---------------------------------------------------------------------------
// Pre-pass B: batched [K][N] f32 -> [N][K] bf16 transpose-convert.
// grid = (K/64, N/64, batch); 64x64 tile via padded LDS.
// ---------------------------------------------------------------------------
__global__ __launch_bounds__(TPB) void transpose_convert_kernel(
    const float* __restrict__ in, bf16_t* __restrict__ out, int K, int N) {
    __shared__ bf16_t tile[64][65];       // pad -> conflict-free transpose
    const int k0 = blockIdx.x * 64;
    const int n0 = blockIdx.y * 64;
    const size_t moff = (size_t)blockIdx.z * K * N;
    const float* src = in + moff;
    bf16_t* dst = out + moff;
    const int tid = threadIdx.x;
#pragma unroll
    for (int j = 0; j < 16; ++j) {
        int i = tid + j * TPB;            // 0..4095
        int r = i >> 6, c = i & 63;
        tile[r][c] = (bf16_t)src[(size_t)(k0 + r) * N + (n0 + c)];
    }
    __syncthreads();
#pragma unroll
    for (int j = 0; j < 16; ++j) {
        int i = tid + j * TPB;
        int n = i >> 6, k = i & 63;
        dst[(size_t)(n0 + n) * K + (k0 + k)] = tile[k][n];
    }
}

// ---------------------------------------------------------------------------
// Kernel 1: experts.  grid = (E=100, B/64).  bf16 pre-transposed weights.
// Register double-buffered staging.
// ---------------------------------------------------------------------------
__global__ __launch_bounds__(TPB) void expert_kernel(
    const bf16_t* __restrict__ xb,    // [2048, 512] bf16
    const bf16_t* __restrict__ W1t,   // [100][256][512] bf16 (N-major)
    const float*  __restrict__ b1,    // [100, 256]
    const bf16_t* __restrict__ W2t,   // [100][128][256] bf16 (N-major)
    const float*  __restrict__ b2,    // [100, 128]
    bf16_t* __restrict__ concat)      // [2048, 12800] bf16
{
    constexpr int DIN = 512, H = 256, DOUT = 128, BM = 64, KC = 32, ECOLS = 12800;

    __shared__ alignas(16) bf16_t h_lds[BM * H];             // 32 KB
    __shared__ alignas(16) bf16_t stage[BM * KC + H * KC];   // 20 KB (reused)
    bf16_t* xA   = stage;             // [BM][KC]
    bf16_t* w1s  = stage + BM * KC;   // [H][KC]
    bf16_t* w2s  = stage;             // stage 2: [DOUT][64]
    bf16_t* outT = stage;             // stage 2 epilogue: [BM][DOUT]

    const int e    = blockIdx.x;
    const int mb   = blockIdx.y * BM;
    const int tid  = threadIdx.x;
    const int wave = tid >> 5;
    const int lane = tid & 31;
    const int col  = lane & 15, rb = (lane >> 4) * 8;

    // ---- stage 1: h = relu(x @ W1) ------------------------------------------
    const int mtile = (wave & 3) * 16;
    const int nbase = (wave >> 2) * 128;
    v8f acc[8];
#pragma unroll
    for (int t = 0; t < 8; ++t) acc[t] = vzero8();

    const bf16_t* W1te = W1t + (size_t)e * H * DIN;
    const int sr = tid >> 2, sc8 = (tid & 3) << 3;       // staging coords (x & W1)

    // prologue: prefetch chunk 0 into registers
    v8bf xreg = *(const v8bf*)(xb + (size_t)(mb + sr) * DIN + sc8);
    v8bf wreg[4];
#pragma unroll
    for (int j = 0; j < 4; ++j) {
        int i = tid + j * TPB;
        wreg[j] = *(const v8bf*)(W1te + (size_t)(i >> 2) * DIN + ((i & 3) << 3));
    }

    for (int k0 = 0; k0 < DIN; k0 += KC) {
        __syncthreads();   // previous chunk fully consumed
        *(v8bf*)(xA + sr * KC + sc8) = xreg;
#pragma unroll
        for (int j = 0; j < 4; ++j) {
            int i = tid + j * TPB;
            *(v8bf*)(w1s + (i >> 2) * KC + ((i & 3) << 3)) = wreg[j];
        }
        if (k0 + KC < DIN) {   // prefetch next chunk during compute
            xreg = *(const v8bf*)(xb + (size_t)(mb + sr) * DIN + k0 + KC + sc8);
#pragma unroll
            for (int j = 0; j < 4; ++j) {
                int i = tid + j * TPB;
                wreg[j] = *(const v8bf*)(W1te + (size_t)(i >> 2) * DIN + k0 + KC + ((i & 3) << 3));
            }
        }
        __syncthreads();
        v16bf a = load_a_frag(xA + mtile * KC, KC);
#pragma unroll
        for (int t = 0; t < 8; ++t) {
            v16bf b = load_b_frag(w1s + (nbase + t * 16) * KC, KC);
            acc[t] = wmma_bf16(a, b, acc[t]);
        }
    }
    __syncthreads();
    {   // epilogue: bias + relu -> h_lds
        const float* b1e = b1 + e * H;
#pragma unroll
        for (int t = 0; t < 8; ++t) {
            const int n = nbase + t * 16 + col;
            const float bias = b1e[n];
#pragma unroll
            for (int r = 0; r < 8; ++r) {
                float v = acc[t][r] + bias;
                v = v > 0.0f ? v : 0.0f;
                h_lds[(mtile + rb + r) * H + n] = (bf16_t)v;
            }
        }
    }
    __syncthreads();

    // ---- stage 2: out = h @ W2 ----------------------------------------------
    const int n4 = (wave >> 2) * 64;
    v8f acc2[4];
#pragma unroll
    for (int t = 0; t < 4; ++t) acc2[t] = vzero8();

    const bf16_t* W2te = W2t + (size_t)e * DOUT * H;
    v8bf w2reg[4];
#pragma unroll
    for (int j = 0; j < 4; ++j) {
        int i = tid + j * TPB;
        w2reg[j] = *(const v8bf*)(W2te + (size_t)(i >> 3) * H + ((i & 7) << 3));
    }

    for (int k0 = 0; k0 < H; k0 += 64) {
        __syncthreads();
#pragma unroll
        for (int j = 0; j < 4; ++j) {
            int i = tid + j * TPB;
            *(v8bf*)(w2s + (i >> 3) * 64 + ((i & 7) << 3)) = w2reg[j];
        }
        if (k0 + 64 < H) {
#pragma unroll
            for (int j = 0; j < 4; ++j) {
                int i = tid + j * TPB;
                w2reg[j] = *(const v8bf*)(W2te + (size_t)(i >> 3) * H + k0 + 64 + ((i & 7) << 3));
            }
        }
        __syncthreads();
#pragma unroll
        for (int ks = 0; ks < 64; ks += 32) {
            v16bf a = load_a_frag(h_lds + mtile * H + k0 + ks, H);
#pragma unroll
            for (int t = 0; t < 4; ++t) {
                v16bf b = load_b_frag(w2s + (n4 + t * 16) * 64 + ks, 64);
                acc2[t] = wmma_bf16(a, b, acc2[t]);
            }
        }
    }
    __syncthreads();
    {   // epilogue: bias -> LDS tile, then 16B vector stores to concat
        const float* b2e = b2 + e * DOUT;
#pragma unroll
        for (int t = 0; t < 4; ++t) {
            const int n = n4 + t * 16 + col;
            const float bias = b2e[n];
#pragma unroll
            for (int r = 0; r < 8; ++r)
                outT[(mtile + rb + r) * DOUT + n] = (bf16_t)(acc2[t][r] + bias);
        }
        __syncthreads();
        bf16_t* cdst = concat + (size_t)mb * ECOLS + (size_t)e * DOUT;
#pragma unroll
        for (int j = 0; j < 4; ++j) {  // 64x128 bf16, 1024 x 16B stores
            int i = tid + j * TPB;
            int r = i >> 4, c8 = (i & 15) << 3;
            *(v8bf*)(cdst + (size_t)r * ECOLS + c8) = *(const v8bf*)(outT + r * DOUT + c8);
        }
    }
}

// ---------------------------------------------------------------------------
// Kernel 2: y = relu(concat[2048,12800] @ Wc1 + bc1) -> bf16
// grid = (B/64, 512/128).  Async global->LDS copies (ASYNCcnt) with LDS
// ping-pong: one barrier per K-chunk; copy of chunk k+1 overlaps WMMA of k.
// Falls back to register staging if the async builtins are unavailable.
// ---------------------------------------------------------------------------
__global__ __launch_bounds__(TPB) void consensus1_kernel(
    const bf16_t* __restrict__ concat,  // [2048, 12800] bf16
    const bf16_t* __restrict__ Wc1t,    // [512][12800] bf16 (N-major)
    const float*  __restrict__ bc1,     // [512]
    bf16_t* __restrict__ y)             // [2048, 512] bf16 (post-relu)
{
    constexpr int K = 12800, N = 512, BM = 64, BN = 128, KC = 64;
    __shared__ alignas(16) bf16_t aA[2][BM * KC];   // 2 x  8 KB
    __shared__ alignas(16) bf16_t wt[2][BN * KC];   // 2 x 16 KB (wt[0] reused as out tile)

    const int mb   = blockIdx.x * BM;
    const int nb   = blockIdx.y * BN;
    const int tid  = threadIdx.x;
    const int wave = tid >> 5;
    const int lane = tid & 31;
    const int col  = lane & 15, rb = (lane >> 4) * 8;

    const int mtile = (wave & 3) * 16;
    const int ng    = (wave >> 2) * 64;
    v8f acc[4];
#pragma unroll
    for (int t = 0; t < 4; ++t) acc[t] = vzero8();

    const bf16_t* wsrc = Wc1t + (size_t)nb * K;

    // issue all 6 async 16B copies for one chunk into buffer `buf`
    auto stage_chunk = [&](int buf, int k0) {
#pragma unroll
        for (int j = 0; j < 2; ++j) {
            int i = tid + j * TPB;
            int r = i >> 3, c8 = (i & 7) << 3;
            async_cp16(&aA[buf][r * KC + c8],
                       concat + (size_t)(mb + r) * K + k0 + c8);
        }
#pragma unroll
        for (int j = 0; j < 4; ++j) {
            int i = tid + j * TPB;
            int n = i >> 3, c8 = (i & 7) << 3;
            async_cp16(&wt[buf][n * KC + c8],
                       wsrc + (size_t)n * K + k0 + c8);
        }
    };

    stage_chunk(0, 0);                   // prologue: chunk 0 -> buffer 0
    int p = 0;
    for (int k0 = 0; k0 < K; k0 += KC, p ^= 1) {
        wait_async();                    // my async copies for buffer p done
        __syncthreads();                 // everyone's copies done; prev compute done
        if (k0 + KC < K)
            stage_chunk(p ^ 1, k0 + KC); // overlap next copy with this compute
#pragma unroll
        for (int ks = 0; ks < KC; ks += 32) {
            v16bf a = load_a_frag(&aA[p][mtile * KC + ks], KC);
#pragma unroll
            for (int t = 0; t < 4; ++t) {
                v16bf b = load_b_frag(&wt[p][(ng + t * 16) * KC + ks], KC);
                acc[t] = wmma_bf16(a, b, acc[t]);
            }
        }
    }
    __syncthreads();
    {   // epilogue: bias + relu -> LDS tile (wt[0]) -> 16B vector stores to y
#pragma unroll
        for (int t = 0; t < 4; ++t) {
            const int n = ng + t * 16 + col;
            const float bias = bc1[nb + n];
#pragma unroll
            for (int r = 0; r < 8; ++r) {
                float v = acc[t][r] + bias;
                v = v > 0.0f ? v : 0.0f;
                wt[0][(mtile + rb + r) * BN + n] = (bf16_t)v;   // [64][128] tile
            }
        }
        __syncthreads();
#pragma unroll
        for (int j = 0; j < 4; ++j) {  // 64x128, 1024 x 16B stores
            int i = tid + j * TPB;
            int r = i >> 4, c8 = (i & 15) << 3;
            *(v8bf*)(y + (size_t)(mb + r) * N + nb + c8) = *(const v8bf*)(&wt[0][r * BN + c8]);
        }
    }
}

// ---------------------------------------------------------------------------
// Kernel 3: out = ((y @ Wc2 + bc2) @ We + be) @ Wd + bd ; intermediates in LDS.
// grid = B/64.  All weights pre-transposed bf16.
// ---------------------------------------------------------------------------
__global__ __launch_bounds__(TPB) void tail_kernel(
    const bf16_t* __restrict__ y,     // [2048, 512] bf16
    const bf16_t* __restrict__ Wc2t,  // [128][512] bf16
    const float*  __restrict__ bc2,
    const bf16_t* __restrict__ Wet,   // [128][128] bf16
    const float*  __restrict__ be,
    const bf16_t* __restrict__ Wdt,   // [128][128] bf16
    const float*  __restrict__ bd,
    float* __restrict__ out)          // [2048, 128]
{
    constexpr int BM = 64, N = 128, KC = 64, KY = 512;
    __shared__ alignas(16) bf16_t aA[BM * KC];     //  8 KB
    __shared__ alignas(16) bf16_t wt[N * KC];      // 16 KB
    __shared__ alignas(16) bf16_t z_lds[BM * N];   // 16 KB
    __shared__ alignas(16) bf16_t e_lds[BM * N];   // 16 KB

    const int mb   = blockIdx.x * BM;
    const int tid  = threadIdx.x;
    const int wave = tid >> 5;
    const int lane = tid & 31;
    const int mtile = (wave & 3) * 16;
    const int ng    = (wave >> 2) * 64;
    const int col = lane & 15, rb = (lane >> 4) * 8;

    v8f acc[4];

    // ---- phase A: z = y @ Wc2 + bc2 ----------------------------------------
#pragma unroll
    for (int t = 0; t < 4; ++t) acc[t] = vzero8();
    for (int k0 = 0; k0 < KY; k0 += KC) {
#pragma unroll
        for (int j = 0; j < 2; ++j) {
            int i = tid + j * TPB;
            *(v8bf*)(aA + (i >> 3) * KC + ((i & 7) << 3)) =
                *(const v8bf*)(y + (size_t)(mb + (i >> 3)) * KY + k0 + ((i & 7) << 3));
        }
#pragma unroll
        for (int j = 0; j < 4; ++j) {
            int i = tid + j * TPB;
            *(v8bf*)(wt + (i >> 3) * KC + ((i & 7) << 3)) =
                *(const v8bf*)(Wc2t + (size_t)(i >> 3) * KY + k0 + ((i & 7) << 3));
        }
        __syncthreads();
#pragma unroll
        for (int ks = 0; ks < KC; ks += 32) {
            v16bf a = load_a_frag(aA + mtile * KC + ks, KC);
#pragma unroll
            for (int t = 0; t < 4; ++t) {
                v16bf b = load_b_frag(wt + (ng + t * 16) * KC + ks, KC);
                acc[t] = wmma_bf16(a, b, acc[t]);
            }
        }
        __syncthreads();
    }
#pragma unroll
    for (int t = 0; t < 4; ++t) {
        const int n = ng + t * 16 + col;
        const float bias = bc2[n];
#pragma unroll
        for (int r = 0; r < 8; ++r)
            z_lds[(mtile + rb + r) * N + n] = (bf16_t)(acc[t][r] + bias);
    }
    __syncthreads();

    // ---- phase B: e = z @ We + be ------------------------------------------
#pragma unroll
    for (int t = 0; t < 4; ++t) acc[t] = vzero8();
    for (int k0 = 0; k0 < N; k0 += KC) {
#pragma unroll
        for (int j = 0; j < 4; ++j) {
            int i = tid + j * TPB;
            *(v8bf*)(wt + (i >> 3) * KC + ((i & 7) << 3)) =
                *(const v8bf*)(Wet + (size_t)(i >> 3) * N + k0 + ((i & 7) << 3));
        }
        __syncthreads();
#pragma unroll
        for (int ks = 0; ks < KC; ks += 32) {
            v16bf a = load_a_frag(z_lds + mtile * N + k0 + ks, N);
#pragma unroll
            for (int t = 0; t < 4; ++t) {
                v16bf b = load_b_frag(wt + (ng + t * 16) * KC + ks, KC);
                acc[t] = wmma_bf16(a, b, acc[t]);
            }
        }
        __syncthreads();
    }
#pragma unroll
    for (int t = 0; t < 4; ++t) {
        const int n = ng + t * 16 + col;
        const float bias = be[n];
#pragma unroll
        for (int r = 0; r < 8; ++r)
            e_lds[(mtile + rb + r) * N + n] = (bf16_t)(acc[t][r] + bias);
    }
    __syncthreads();

    // ---- phase C: out = e @ Wd + bd ----------------------------------------
#pragma unroll
    for (int t = 0; t < 4; ++t) acc[t] = vzero8();
    for (int k0 = 0; k0 < N; k0 += KC) {
#pragma unroll
        for (int j = 0; j < 4; ++j) {
            int i = tid + j * TPB;
            *(v8bf*)(wt + (i >> 3) * KC + ((i & 7) << 3)) =
                *(const v8bf*)(Wdt + (size_t)(i >> 3) * N + k0 + ((i & 7) << 3));
        }
        __syncthreads();
#pragma unroll
        for (int ks = 0; ks < KC; ks += 32) {
            v16bf a = load_a_frag(e_lds + mtile * N + k0 + ks, N);
#pragma unroll
            for (int t = 0; t < 4; ++t) {
                v16bf b = load_b_frag(wt + (ng + t * 16) * KC + ks, KC);
                acc[t] = wmma_bf16(a, b, acc[t]);
            }
        }
        __syncthreads();
    }
#pragma unroll
    for (int t = 0; t < 4; ++t) {
        const int n = ng + t * 16 + col;
        const float bias = bd[n];
#pragma unroll
        for (int r = 0; r < 8; ++r)
            out[(size_t)(mb + mtile + rb + r) * N + n] = acc[t][r] + bias;
    }
}

// ---------------------------------------------------------------------------
extern "C" void kernel_launch(void* const* d_in, const int* in_sizes, int n_in,
                              void* d_out, int out_size, void* d_ws, size_t ws_size,
                              hipStream_t stream) {
    (void)in_sizes; (void)n_in; (void)out_size; (void)ws_size;

    const float* x   = (const float*)d_in[0];
    const float* W1  = (const float*)d_in[1];
    const float* b1  = (const float*)d_in[2];
    const float* W2  = (const float*)d_in[3];
    const float* b2  = (const float*)d_in[4];
    const float* Wc1 = (const float*)d_in[5];
    const float* bc1 = (const float*)d_in[6];
    const float* Wc2 = (const float*)d_in[7];
    const float* bc2 = (const float*)d_in[8];
    const float* We  = (const float*)d_in[9];
    const float* be  = (const float*)d_in[10];
    const float* Wd  = (const float*)d_in[11];
    const float* bd  = (const float*)d_in[12];
    float* out = (float*)d_out;

    // ---- workspace layout (bf16 elements) ----------------------------------
    char* ws = (char*)d_ws;
    size_t off = 0;
    bf16_t* concat = (bf16_t*)(ws + off); off += (size_t)2048 * 12800 * 2;  // 52.4 MB
    bf16_t* yb     = (bf16_t*)(ws + off); off += (size_t)2048 * 512   * 2;  //  2.1 MB
    bf16_t* xb     = (bf16_t*)(ws + off); off += (size_t)2048 * 512   * 2;  //  2.1 MB
    bf16_t* W1t    = (bf16_t*)(ws + off); off += (size_t)100 * 512 * 256 * 2; // 26.2 MB
    bf16_t* W2t    = (bf16_t*)(ws + off); off += (size_t)100 * 256 * 128 * 2; //  6.6 MB
    bf16_t* Wc1t   = (bf16_t*)(ws + off); off += (size_t)12800 * 512 * 2;   // 13.1 MB
    bf16_t* Wc2t   = (bf16_t*)(ws + off); off += (size_t)512 * 128 * 2;
    bf16_t* Wet    = (bf16_t*)(ws + off); off += (size_t)128 * 128 * 2;
    bf16_t* Wdt    = (bf16_t*)(ws + off); off += (size_t)128 * 128 * 2;

    // ---- pre-pass: bf16 conversion + weight transposes ---------------------
    convert_kernel<<<(2048 * 512 / 4 + TPB - 1) / TPB, TPB, 0, stream>>>(
        x, xb, 2048 * 512 / 4);
    transpose_convert_kernel<<<dim3(512 / 64, 256 / 64, 100), TPB, 0, stream>>>(
        W1, W1t, 512, 256);
    transpose_convert_kernel<<<dim3(256 / 64, 128 / 64, 100), TPB, 0, stream>>>(
        W2, W2t, 256, 128);
    transpose_convert_kernel<<<dim3(12800 / 64, 512 / 64, 1), TPB, 0, stream>>>(
        Wc1, Wc1t, 12800, 512);
    transpose_convert_kernel<<<dim3(512 / 64, 128 / 64, 1), TPB, 0, stream>>>(
        Wc2, Wc2t, 512, 128);
    transpose_convert_kernel<<<dim3(128 / 64, 128 / 64, 1), TPB, 0, stream>>>(
        We, Wet, 128, 128);
    transpose_convert_kernel<<<dim3(128 / 64, 128 / 64, 1), TPB, 0, stream>>>(
        Wd, Wdt, 128, 128);

    // ---- main pipeline ------------------------------------------------------
    dim3 g1(100, 2048 / 64);
    expert_kernel<<<g1, TPB, 0, stream>>>(xb, W1t, b1, W2t, b2, concat);

    dim3 g2(2048 / 64, 512 / 128);
    consensus1_kernel<<<g2, TPB, 0, stream>>>(concat, Wc1t, bc1, yb);

    tail_kernel<<<2048 / 64, TPB, 0, stream>>>(
        yb, Wc2t, bc2, Wet, be, Wdt, bd, out);
}
